// LGNNPlusRAT_53223234732415
// MI455X (gfx1250) — compile-verified, matrix operand order checked by reading, not run on hardware
//
#include <hip/hip_runtime.h>
#include <hip/hip_bf16.h>

// ---------------------------------------------------------------------------
// CDNA5 / gfx1250 implementation of LGNNPlusRAT (4-layer node + line-graph
// transformer). Dense GEMMs (~355 GFLOP) run on v_wmma_f32_16x16x32_bf16
// (f32 -> bf16 LDS tiles, f32 accumulate); each wave computes a 32x64 strip
// (8 WMMAs per 32-K step, B fragments reused across both A fragments).
// Sparse attention segment-softmax (~7 GB irregular traffic, memory bound at
// 23.3 TB/s) uses VALU + f32 global atomics. GEMM fuses row-gather on A,
// bias, residual add and ReLU; global_prefetch_b8 hides next-tile latency.
// ---------------------------------------------------------------------------

typedef __attribute__((ext_vector_type(16))) __bf16 v16bf;
typedef __attribute__((ext_vector_type(8)))  __bf16 v8bf;
typedef __attribute__((ext_vector_type(8)))  float  v8f;

__device__ __forceinline__ v16bf load_frag(const __bf16* base, int kh)
{
    // ISA 16-bit fragment layout: lane<16 K {0..7,16..23}, lane>=16 K {8..15,24..31}
    v16bf f;
    v8bf lo = *(const v8bf*)(base + kh * 8);
    v8bf hi = *(const v8bf*)(base + 16 + kh * 8);
    #pragma unroll
    for (int i = 0; i < 8; ++i) { f[i] = lo[i]; f[8 + i] = hi[i]; }
    return f;
}

// C[M,Nc] = act( A[aidx[row]][:] @ B + bias + res );  Nc%64==0, K%32==0, lda==K
// block = 128 threads (4 waves); tile 128x64; per-wave 32x64 (2 A frags x 4 B frags)
__launch_bounds__(128)
__global__ void gemm_bf16_wmma(const float* __restrict__ A, const int* __restrict__ aidx,
                               const float* __restrict__ B, const float* __restrict__ bias,
                               const float* res, float* C,
                               int M, int Nc, int K, int relu)
{
    constexpr int LDT = 40;            // bf16 per LDS row (80B -> 16B-aligned runs)
    __shared__ __attribute__((aligned(16))) __bf16 As[128 * LDT];
    __shared__ __attribute__((aligned(16))) __bf16 Bs[64 * LDT];

    const int tid  = threadIdx.x;
    const int wave = tid >> 5;
    const int lane = tid & 31;
    const int r    = lane & 15;
    const int kh   = lane >> 4;        // K-half selector (ISA 16-bit layout)
    const int row0 = blockIdx.y * 128;
    const int col0 = blockIdx.x * 64;

    v8f acc[2][4] = {};

    // A staging: one thread per tile row (128 rows x 32 cols)
    const int agr = row0 + tid;
    int arow = -1;
    if (agr < M) arow = aidx ? aidx[agr] : agr;
    // B staging: 4 threads per k-row, 16 cols each
    const int bk_l = tid >> 2;          // 0..31
    const int bc_l = (tid & 3) * 16;    // 0,16,32,48

    for (int k0 = 0; k0 < K; k0 += 32) {
        {   // stage A tile f32 -> bf16
            float vals[32];
            if (arow >= 0) {
                const float* src = A + (size_t)arow * K + k0;
                #pragma unroll
                for (int j = 0; j < 32; ++j) vals[j] = src[j];
                if (k0 + 32 < K) __builtin_prefetch(src + 32, 0, 0);
            } else {
                #pragma unroll
                for (int j = 0; j < 32; ++j) vals[j] = 0.0f;
            }
            #pragma unroll
            for (int j = 0; j < 32; ++j) As[tid * LDT + j] = (__bf16)vals[j];
        }
        {   // stage B tile (32x64) transposed: Bs[n][k]
            const float* src = B + (size_t)(k0 + bk_l) * Nc + (size_t)(col0 + bc_l);
            #pragma unroll
            for (int j = 0; j < 16; ++j) Bs[(bc_l + j) * LDT + bk_l] = (__bf16)src[j];
            if (k0 + 32 < K) __builtin_prefetch(src + (size_t)32 * Nc, 0, 0);
        }
        __syncthreads();

        // issue ALL fragment loads first, then 8 back-to-back WMMAs
        v16bf a0 = load_frag(&As[(wave * 32 + r) * LDT], kh);
        v16bf a1 = load_frag(&As[(wave * 32 + 16 + r) * LDT], kh);
        v16bf bf0 = load_frag(&Bs[(0 * 16 + r) * LDT], kh);
        v16bf bf1 = load_frag(&Bs[(1 * 16 + r) * LDT], kh);
        v16bf bf2 = load_frag(&Bs[(2 * 16 + r) * LDT], kh);
        v16bf bf3 = load_frag(&Bs[(3 * 16 + r) * LDT], kh);

        acc[0][0] = __builtin_amdgcn_wmma_f32_16x16x32_bf16(false, a0, false, bf0, (short)0, acc[0][0], false, false);
        acc[0][1] = __builtin_amdgcn_wmma_f32_16x16x32_bf16(false, a0, false, bf1, (short)0, acc[0][1], false, false);
        acc[0][2] = __builtin_amdgcn_wmma_f32_16x16x32_bf16(false, a0, false, bf2, (short)0, acc[0][2], false, false);
        acc[0][3] = __builtin_amdgcn_wmma_f32_16x16x32_bf16(false, a0, false, bf3, (short)0, acc[0][3], false, false);
        acc[1][0] = __builtin_amdgcn_wmma_f32_16x16x32_bf16(false, a1, false, bf0, (short)0, acc[1][0], false, false);
        acc[1][1] = __builtin_amdgcn_wmma_f32_16x16x32_bf16(false, a1, false, bf1, (short)0, acc[1][1], false, false);
        acc[1][2] = __builtin_amdgcn_wmma_f32_16x16x32_bf16(false, a1, false, bf2, (short)0, acc[1][2], false, false);
        acc[1][3] = __builtin_amdgcn_wmma_f32_16x16x32_bf16(false, a1, false, bf3, (short)0, acc[1][3], false, false);

        __syncthreads();
    }

    // D: VGPR j -> M = j + 8*kh, N = lane&15
    #pragma unroll
    for (int f = 0; f < 2; ++f) {
        #pragma unroll
        for (int s = 0; s < 4; ++s) {
            const int col = col0 + s * 16 + r;
            #pragma unroll
            for (int j = 0; j < 8; ++j) {
                const int row = row0 + wave * 32 + f * 16 + kh * 8 + j;
                if (row < M) {
                    float v = acc[f][s][j];
                    if (bias) v += bias[col];
                    if (res)  v += res[(size_t)row * Nc + col];
                    if (relu) v = fmaxf(v, 0.0f);
                    C[(size_t)row * Nc + col] = v;
                }
            }
        }
    }
}

__launch_bounds__(64)
__global__ void node_attn(const float* __restrict__ q, const float* __restrict__ k,
                          const float* __restrict__ v, const float* __restrict__ lgx,
                          const int* __restrict__ gsrc, const int* __restrict__ gdst,
                          float* wv, float* z)
{
    const int e = blockIdx.x, t = threadIdx.x, wid = t >> 5;
    const int s = gsrc[e], d = gdst[e];
    __shared__ float red[16];
    const float  le = lgx[(size_t)e * 64 + t];
    const float* ks = k + (size_t)s * 512;
    const float* vs = v + (size_t)s * 512;
    const float* qd = q + (size_t)d * 512;
    #pragma unroll
    for (int h = 0; h < 8; ++h) {
        float p = (ks[h * 64 + t] + le) * qd[h * 64 + t];
        #pragma unroll
        for (int off = 16; off > 0; off >>= 1) p += __shfl_down(p, off, 32);
        if ((t & 31) == 0) red[h * 2 + wid] = p;
    }
    __syncthreads();
    #pragma unroll
    for (int h = 0; h < 8; ++h) {
        const float sc = (red[2 * h] + red[2 * h + 1]) * 0.125f;      // 1/sqrt(64)
        const float w  = __expf(fminf(fmaxf(sc, -10.0f), 10.0f));
        atomicAdd(&wv[(size_t)d * 512 + h * 64 + t], w * (vs[h * 64 + t] + le));
    }
    if (t < 8) {
        const float sc = (red[2 * t] + red[2 * t + 1]) * 0.125f;
        const float w  = __expf(fminf(fmaxf(sc, -10.0f), 10.0f));
        atomicAdd(&z[(size_t)d * 8 + t], w);
    }
}

__launch_bounds__(64)
__global__ void edge_attn(const float* __restrict__ qe, const float* __restrict__ ke,
                          const float* __restrict__ ve,
                          const int* __restrict__ lsrc, const int* __restrict__ ldst,
                          float* wve, float* ze)
{
    const int m = blockIdx.x, t = threadIdx.x;   // h = t>>3, d = t&7
    const int s = lsrc[m], d = ldst[m];
    float p = ke[(size_t)s * 64 + t] * qe[(size_t)d * 64 + t];
    p += __shfl_xor(p, 1, 32);
    p += __shfl_xor(p, 2, 32);
    p += __shfl_xor(p, 4, 32);
    const float sc = p * 0.35355339059327373f;   // 1/sqrt(8)
    const float w  = __expf(fminf(fmaxf(sc, -10.0f), 10.0f));
    atomicAdd(&wve[(size_t)d * 64 + t], w * ve[(size_t)s * 64 + t]);
    if ((t & 7) == 0) atomicAdd(&ze[(size_t)d * 8 + (t >> 3)], w);
}

__launch_bounds__(64)
__global__ void layernorm_k(float* out, const float* in, const float* __restrict__ g,
                            const float* __restrict__ b, int D)
{
    const int row = blockIdx.x, t = threadIdx.x;
    const float* x = in + (size_t)row * D;
    float s = 0.f, s2 = 0.f;
    for (int i = t; i < D; i += 64) { float v = x[i]; s += v; s2 += v * v; }
    __shared__ float rs[64], rq[64];
    rs[t] = s; rq[t] = s2;
    __syncthreads();
    #pragma unroll
    for (int off = 32; off > 0; off >>= 1) {
        if (t < off) { rs[t] += rs[t + off]; rq[t] += rq[t + off]; }
        __syncthreads();
    }
    const float mu  = rs[0] / (float)D;
    const float var = rq[0] / (float)D - mu * mu;
    const float inv = rsqrtf(var + 1e-5f);
    float* o = out + (size_t)row * D;
    for (int i = t; i < D; i += 64) o[i] = (x[i] - mu) * inv * g[i] + b[i];
}

__global__ void zero_k(float* p, int n)
{ int i = blockIdx.x * blockDim.x + threadIdx.x; if (i < n) p[i] = 0.0f; }

__global__ void copy_k(float* dst, const float* src, int n)
{ int i = blockIdx.x * blockDim.x + threadIdx.x; if (i < n) dst[i] = src[i]; }

__global__ void gather_rows_k(float* dst, const float* __restrict__ src,
                              const int* __restrict__ idx, int rows, int cols)
{
    int i = blockIdx.x * blockDim.x + threadIdx.x;
    if (i < rows * cols) { int r = i / cols, c = i - r * cols;
        dst[i] = src[(size_t)idx[r] * cols + c]; }
}

__global__ void scatter_rows_k(float* dst, const float* __restrict__ src,
                               const int* __restrict__ idx, int rows, int cols)
{
    int i = blockIdx.x * blockDim.x + threadIdx.x;
    if (i < rows * cols) { int r = i / cols, c = i - r * cols;
        dst[(size_t)idx[r] * cols + c] = src[i]; }
}

__global__ void attn_div_k(float* acc, const float* __restrict__ z, int n, int dk_shift)
{
    int i = blockIdx.x * blockDim.x + threadIdx.x;
    if (i < n) acc[i] /= fmaxf(z[i >> dk_shift], 1e-9f);
}

extern "C" void kernel_launch(void* const* d_in, const int* in_sizes, int n_in,
                              void* d_out, int out_size, void* d_ws, size_t ws_size,
                              hipStream_t stream)
{
    (void)in_sizes; (void)n_in; (void)out_size; (void)ws_size;
    constexpr int N = 10000, D = 512, H = 8, E = 120000, EL = 60000, M = 600000;
    constexpr int ED = 64, EH = 8, L = 4, FD = 4 * D, GD = 4 * ED;

    const float* x0  = (const float*)d_in[0];
    const float* rel = (const float*)d_in[1];
    const float* Wq  = (const float*)d_in[2];  const float* bq  = (const float*)d_in[3];
    const float* Wk  = (const float*)d_in[4];  const float* Wv  = (const float*)d_in[5];
    const float* Wo  = (const float*)d_in[6];  const float* bo  = (const float*)d_in[7];
    const float* l1g = (const float*)d_in[8];  const float* l1b = (const float*)d_in[9];
    const float* F1  = (const float*)d_in[10]; const float* f1b = (const float*)d_in[11];
    const float* F2  = (const float*)d_in[12]; const float* f2b = (const float*)d_in[13];
    const float* l2g = (const float*)d_in[14]; const float* l2b = (const float*)d_in[15];
    const float* Eq  = (const float*)d_in[16]; const float* eqb = (const float*)d_in[17];
    const float* Ek  = (const float*)d_in[18]; const float* Ev  = (const float*)d_in[19];
    const float* Eo  = (const float*)d_in[20]; const float* eob = (const float*)d_in[21];
    const float* Ns  = (const float*)d_in[22]; const float* nsb = (const float*)d_in[23];
    const float* Nd  = (const float*)d_in[24]; const float* ndb = (const float*)d_in[25];
    const float* e1g = (const float*)d_in[26]; const float* e1b = (const float*)d_in[27];
    const float* G1  = (const float*)d_in[28]; const float* g1b = (const float*)d_in[29];
    const float* G2  = (const float*)d_in[30]; const float* g2b = (const float*)d_in[31];
    const float* e2g = (const float*)d_in[32]; const float* e2b = (const float*)d_in[33];
    const int* edge_feat   = (const int*)d_in[34];
    const int* local_index = (const int*)d_in[35];
    const int* src_ids     = (const int*)d_in[36];
    const int* dst_ids     = (const int*)d_in[37];
    const int* g_src       = (const int*)d_in[38];
    const int* g_dst       = (const int*)d_in[39];
    const int* lg_src      = (const int*)d_in[40];
    const int* lg_dst      = (const int*)d_in[41];

    float* ws = (float*)d_ws;
    size_t off = 0;
    auto carve = [&](size_t n) { float* p = ws + off; off += n; return p; };
    float* xA  = carve((size_t)N * D);
    float* xB  = carve((size_t)N * D);
    float* lgx = carve((size_t)E * ED);
    float* llA = carve((size_t)EL * ED);
    float* llB = carve((size_t)EL * ED);
    float* qb  = carve((size_t)N * D);     // node q; reused as qe
    float* kb  = carve((size_t)N * D);     // node k; reused as ke
    float* vb  = carve((size_t)N * D);     // node v; reused as ve
    float* wvb = carve((size_t)N * D);     // node attn acc -> o; reused as me
    float* zb  = carve((size_t)N * H);
    float* hb  = carve((size_t)N * D);
    float* mid = carve((size_t)N * FD);    // FFN mid (node + edge, reused)
    float* wve = carve((size_t)EL * ED);
    float* zeb = carve((size_t)EL * EH);
    float* heb = carve((size_t)EL * ED);

    auto ew = [&](int n) { return dim3((unsigned)((n + 255) / 256)); };
    auto gemm = [&](const float* A, const int* ai, const float* B, const float* bias,
                    const float* res, float* C, int Mm, int Nn, int Kk, int relu) {
        dim3 g((unsigned)(Nn / 64), (unsigned)((Mm + 127) / 128));
        gemm_bf16_wmma<<<g, 128, 0, stream>>>(A, ai, B, bias, res, C, Mm, Nn, Kk, relu);
    };

    copy_k<<<ew(N * D), 256, 0, stream>>>(xA, x0, N * D);
    gather_rows_k<<<ew(E * ED), 256, 0, stream>>>(lgx, rel, edge_feat, E, ED);
    gather_rows_k<<<ew(EL * ED), 256, 0, stream>>>(llA, lgx, local_index, EL, ED);

    float* xc = xA; float* xn = xB;
    float* lc = llA; float* lx = llB;

    for (int i = 0; i < L; ++i) {
        const float* Wq_i = Wq + (size_t)i * D * D;
        const float* Wk_i = Wk + (size_t)i * D * D;
        const float* Wv_i = Wv + (size_t)i * D * D;
        const float* Wo_i = Wo + (size_t)i * D * D;
        const float* F1_i = F1 + (size_t)i * D * FD;
        const float* F2_i = F2 + (size_t)i * FD * D;
        const float* Eq_i = Eq + (size_t)i * ED * ED;
        const float* Ek_i = Ek + (size_t)i * ED * ED;
        const float* Ev_i = Ev + (size_t)i * ED * ED;
        const float* Eo_i = Eo + (size_t)i * ED * ED;
        const float* Ns_i = Ns + (size_t)i * D * ED;
        const float* Nd_i = Nd + (size_t)i * D * ED;
        const float* G1_i = G1 + (size_t)i * ED * GD;
        const float* G2_i = G2 + (size_t)i * GD * ED;

        // -------- node transformer --------
        gemm(xc, nullptr, Wq_i, bq + (size_t)i * D, nullptr, qb, N, D, D, 0);
        gemm(xc, nullptr, Wk_i, nullptr,            nullptr, kb, N, D, D, 0);
        gemm(xc, nullptr, Wv_i, nullptr,            nullptr, vb, N, D, D, 0);
        zero_k<<<ew(N * D), 256, 0, stream>>>(wvb, N * D);
        zero_k<<<ew(N * H), 256, 0, stream>>>(zb, N * H);
        node_attn<<<dim3(E), 64, 0, stream>>>(qb, kb, vb, lgx, g_src, g_dst, wvb, zb);
        attn_div_k<<<ew(N * D), 256, 0, stream>>>(wvb, zb, N * D, 6);
        gemm(wvb, nullptr, Wo_i, bo + (size_t)i * D, xc, hb, N, D, D, 0);
        layernorm_k<<<dim3(N), 64, 0, stream>>>(hb, hb, l1g + (size_t)i * D, l1b + (size_t)i * D, D);
        gemm(hb, nullptr, F1_i, f1b + (size_t)i * FD, nullptr, mid, N, FD, D, 1);
        gemm(mid, nullptr, F2_i, f2b + (size_t)i * D, hb, xn, N, D, FD, 0);
        layernorm_k<<<dim3(N), 64, 0, stream>>>(xn, xn, l2g + (size_t)i * D, l2b + (size_t)i * D, D);

        // -------- line-graph transformer (uses start-of-layer xc) --------
        gemm(xc, src_ids, Ns_i, nsb + (size_t)i * ED, nullptr, qb,  EL, ED, D, 0);
        gemm(xc, dst_ids, Nd_i, ndb + (size_t)i * ED, nullptr, wvb, EL, ED, D, 0);   // me
        gemm(lc, nullptr, Eq_i, eqb + (size_t)i * ED, qb,  qb, EL, ED, ED, 0);       // qe
        gemm(lc, nullptr, Ek_i, nullptr,              wvb, kb, EL, ED, ED, 0);       // ke
        gemm(lc, nullptr, Ev_i, nullptr,              wvb, vb, EL, ED, ED, 0);       // ve
        zero_k<<<ew(EL * ED), 256, 0, stream>>>(wve, EL * ED);
        zero_k<<<ew(EL * EH), 256, 0, stream>>>(zeb, EL * EH);
        edge_attn<<<dim3(M), 64, 0, stream>>>(qb, kb, vb, lg_src, lg_dst, wve, zeb);
        attn_div_k<<<ew(EL * ED), 256, 0, stream>>>(wve, zeb, EL * ED, 3);
        gemm(wve, nullptr, Eo_i, eob + (size_t)i * ED, lc, heb, EL, ED, ED, 0);
        layernorm_k<<<dim3(EL), 64, 0, stream>>>(heb, heb, e1g + (size_t)i * ED, e1b + (size_t)i * ED, ED);
        gemm(heb, nullptr, G1_i, g1b + (size_t)i * GD, nullptr, mid, EL, GD, ED, 1);
        gemm(mid, nullptr, G2_i, g2b + (size_t)i * ED, heb, lx, EL, ED, GD, 0);
        layernorm_k<<<dim3(EL), 64, 0, stream>>>(lx, lx, e2g + (size_t)i * ED, e2b + (size_t)i * ED, ED);
        scatter_rows_k<<<ew(EL * ED), 256, 0, stream>>>(lgx, lx, local_index, EL, ED);

        { float* t = xc; xc = xn; xn = t; }
        { float* t = lc; lc = lx; lx = t; }
    }

    float* out = (float*)d_out;
    copy_k<<<ew(N * D), 256, 0, stream>>>(out, xc, N * D);
    copy_k<<<ew(EL * ED), 256, 0, stream>>>(out + (size_t)N * D, lc, EL * ED);
}
